// AdaptiveSoftmax_11269994185272
// MI455X (gfx1250) — matrix operand order repaired and checked by reading, not compiled, not run on hardware
//
#include <hip/hip_runtime.h>
#include <hip/hip_bf16.h>
#include <math.h>

// ---------------------------------------------------------------------------
// Adaptive softmax NLL for MI455X (gfx1250, wave32, WMMA bf16 16x16x32).
// f32 global -> regs -> bf16 LDS (double-buffered) -> v_wmma (f32 accumulate).
// Softmax stats computed online in registers; logits never materialized.
// WG = 128 threads (4 waves); WG tile 128 rows x 64 cols; wave = 32x64.
// ---------------------------------------------------------------------------

typedef __bf16 bf16_t;
typedef __attribute__((ext_vector_type(16))) __bf16 v16bf;
typedef __attribute__((ext_vector_type(4)))  __bf16 v4bf;
typedef __attribute__((ext_vector_type(8)))  float  v8f;

#define CUT0 4000
#define CUT1 20000
#define CUT2 50000
#define NROWS 4096
#define NEG_BIG (-1.0e30f)

#define A_TILE (128 * 32)
#define B_TILE (64 * 32)

__device__ __forceinline__ float laneMax16(float v) {
  v = fmaxf(v, __shfl_xor(v, 1, 32));
  v = fmaxf(v, __shfl_xor(v, 2, 32));
  v = fmaxf(v, __shfl_xor(v, 4, 32));
  v = fmaxf(v, __shfl_xor(v, 8, 32));
  return v;
}
__device__ __forceinline__ float laneSum16(float v) {
  v += __shfl_xor(v, 1, 32);
  v += __shfl_xor(v, 2, 32);
  v += __shfl_xor(v, 4, 32);
  v += __shfl_xor(v, 8, 32);
  return v;
}

// Per-thread staging registers for one 128x32 A tile + 32x64 B tile.
struct StageRegs {
  float4 a[8];  // one A row (32 f32)
  float4 b[4];  // 4x4 (k x n) B sub-block
};

template <bool FULL>
__device__ __forceinline__ void load_regs(StageRegs& r,
                                          const float* __restrict__ A,
                                          const float* __restrict__ B,
                                          int rowBase, int colBase, int k0,
                                          int K, int C, int tid) {
  const float4* asrc = (const float4*)(A + (size_t)(rowBase + tid) * K + k0);
#pragma unroll
  for (int j = 0; j < 8; ++j) r.a[j] = asrc[j];
  const int kk = (tid & 7) * 4;   // 0..28
  const int nn = (tid >> 3) * 4;  // 0..60
#pragma unroll
  for (int i = 0; i < 4; ++i) {
    const float* p = B + (size_t)(k0 + kk + i) * C + colBase + nn;
    if (FULL) {
      // 8B-aligned even for C=4002 (row stride 16008 B)
      float2 lo = *(const float2*)(p);
      float2 hi = *(const float2*)(p + 2);
      r.b[i] = make_float4(lo.x, lo.y, hi.x, hi.y);
    } else {
      float x0 = (colBase + nn + 0 < C) ? p[0] : 0.0f;
      float x1 = (colBase + nn + 1 < C) ? p[1] : 0.0f;
      float x2 = (colBase + nn + 2 < C) ? p[2] : 0.0f;
      float x3 = (colBase + nn + 3 < C) ? p[3] : 0.0f;
      r.b[i] = make_float4(x0, x1, x2, x3);
    }
  }
}

__device__ __forceinline__ void store_lds(const StageRegs& r,
                                          bf16_t* __restrict__ Asm,
                                          bf16_t* __restrict__ Bsm, int tid) {
  bf16_t* dst = &Asm[tid * 32];
#pragma unroll
  for (int j = 0; j < 8; ++j) {
    dst[4 * j + 0] = (bf16_t)r.a[j].x;
    dst[4 * j + 1] = (bf16_t)r.a[j].y;
    dst[4 * j + 2] = (bf16_t)r.a[j].z;
    dst[4 * j + 3] = (bf16_t)r.a[j].w;
  }
  const int kk = (tid & 7) * 4;
  const int nn = (tid >> 3) * 4;
  // register transpose: per n-row, 4 contiguous k values -> one 8B LDS store
  {
    v4bf p; p[0] = (bf16_t)r.b[0].x; p[1] = (bf16_t)r.b[1].x; p[2] = (bf16_t)r.b[2].x; p[3] = (bf16_t)r.b[3].x;
    *(v4bf*)(&Bsm[(nn + 0) * 32 + kk]) = p;
  }
  {
    v4bf p; p[0] = (bf16_t)r.b[0].y; p[1] = (bf16_t)r.b[1].y; p[2] = (bf16_t)r.b[2].y; p[3] = (bf16_t)r.b[3].y;
    *(v4bf*)(&Bsm[(nn + 1) * 32 + kk]) = p;
  }
  {
    v4bf p; p[0] = (bf16_t)r.b[0].z; p[1] = (bf16_t)r.b[1].z; p[2] = (bf16_t)r.b[2].z; p[3] = (bf16_t)r.b[3].z;
    *(v4bf*)(&Bsm[(nn + 2) * 32 + kk]) = p;
  }
  {
    v4bf p; p[0] = (bf16_t)r.b[0].w; p[1] = (bf16_t)r.b[1].w; p[2] = (bf16_t)r.b[2].w; p[3] = (bf16_t)r.b[3].w;
    *(v4bf*)(&Bsm[(nn + 3) * 32 + kk]) = p;
  }
}

// A fragment (16 rows at waveRow), contiguous k order used here (B matches):
// lanes 0-15: K in [0,16); lanes 16-31: K in [16,32).
__device__ __forceinline__ v16bf load_afrag(const bf16_t* Asm, int waveRow,
                                            int l15, int half) {
  const bf16_t* p0 = &Asm[(waveRow + l15) * 32 + half * 16];
  const bf16_t* p1 = &Asm[(waveRow + l15) * 32 + half * 16 + 8];
  v16bf a;
#pragma unroll
  for (int j = 0; j < 8; ++j) { a[j] = p0[j]; a[8 + j] = p1[j]; }
  return a;
}

// B fragment (16 cols at ctile); LDS is [n][k] so one contiguous 32B read.
__device__ __forceinline__ v16bf load_bfrag(const bf16_t* Bsm, int ctile,
                                            int l15, int half) {
  return *(const v16bf*)(&Bsm[(ctile + l15) * 32 + half * 16]);
}

#define WMMA_BF16(a, b, c) \
  __builtin_amdgcn_wmma_f32_16x16x32_bf16(false, (a), false, (b), (short)0, (c), false, false)

__device__ __forceinline__ void compute_tile(const bf16_t* Asm, const bf16_t* Bsm,
                                             int wave, int l15, int half,
                                             v8f acc[2][4]) {
  v16bf a0 = load_afrag(Asm, wave * 32, l15, half);
  v16bf a1 = load_afrag(Asm, wave * 32 + 16, l15, half);
#pragma unroll
  for (int t = 0; t < 4; ++t) {
    v16bf b = load_bfrag(Bsm, t * 16, l15, half);
    acc[0][t] = WMMA_BF16(a0, b, acc[0][t]);
    acc[1][t] = WMMA_BF16(a1, b, acc[1][t]);
  }
}

// Double-buffered K loop: prefetch tile k+1 into regs before computing tile k.
// One barrier per iteration; 2 LDS buffers.
template <bool FULL>
__device__ __forceinline__ void kloop(const float* __restrict__ A,
                                      const float* __restrict__ B,
                                      bf16_t* __restrict__ Asm,  // [2][A_TILE]
                                      bf16_t* __restrict__ Bsm,  // [2][B_TILE]
                                      int rowBase, int colBase, int K, int C,
                                      int tid, int wave, int l15, int half,
                                      v8f acc[2][4]) {
  StageRegs r;
  load_regs<FULL>(r, A, B, rowBase, colBase, 0, K, C, tid);
  store_lds(r, Asm, Bsm, tid);
  __syncthreads();
  int cur = 0;
  for (int k0 = 0; k0 < K; k0 += 32) {
    const bool more = (k0 + 32) < K;
    if (more) load_regs<FULL>(r, A, B, rowBase, colBase, k0 + 32, K, C, tid);
    compute_tile(Asm + cur * A_TILE, Bsm + cur * B_TILE, wave, l15, half, acc);
    if (more) store_lds(r, Asm + (cur ^ 1) * A_TILE, Bsm + (cur ^ 1) * B_TILE, tid);
    __syncthreads();
    cur ^= 1;
  }
}

// ---------------------------------------------------------------------------
// GEMM: C[f32, MxN] = A[f32, MxK] @ B[f32, KxN]
// ---------------------------------------------------------------------------
__global__ void __launch_bounds__(128)
gemm_bf16(const float* __restrict__ A, const float* __restrict__ B,
          float* __restrict__ C, int M, int K, int N) {
  __shared__ alignas(64) bf16_t Asm[2][A_TILE];
  __shared__ alignas(64) bf16_t Bsm[2][B_TILE];
  const int tid  = threadIdx.x;
  const int lane = tid & 31;
  const int wave = tid >> 5;
  const int l15  = lane & 15;
  const int half = lane >> 4;
  const int rowBase = blockIdx.y * 128;
  const int colBase = blockIdx.x * 64;

  v8f acc[2][4];
#pragma unroll
  for (int r = 0; r < 2; ++r)
#pragma unroll
    for (int t = 0; t < 4; ++t)
#pragma unroll
      for (int e = 0; e < 8; ++e) acc[r][t][e] = 0.0f;

  if (colBase + 64 <= N)
    kloop<true>(A, B, &Asm[0][0], &Bsm[0][0], rowBase, colBase, K, N, tid, wave, l15, half, acc);
  else
    kloop<false>(A, B, &Asm[0][0], &Bsm[0][0], rowBase, colBase, K, N, tid, wave, l15, half, acc);

#pragma unroll
  for (int t = 0; t < 4; ++t) {
    int col = colBase + t * 16 + l15;
    if (col < N) {
#pragma unroll
      for (int r = 0; r < 2; ++r)
#pragma unroll
        for (int v = 0; v < 8; ++v) {
          int row = rowBase + wave * 32 + r * 16 + v + 8 * half;
          C[(size_t)row * N + col] = acc[r][t][v];
        }
    }
  }
}

// ---------------------------------------------------------------------------
// Streaming logits + online softmax stats. WG: 128 rows x one 1024-col chunk.
// ---------------------------------------------------------------------------
__global__ void __launch_bounds__(128)
logits_stats(const float* __restrict__ A, const float* __restrict__ B,
             const float* __restrict__ bias, float* __restrict__ stats,
             int K, int C) {
  __shared__ alignas(64) bf16_t Asm[2][A_TILE];
  __shared__ alignas(64) bf16_t Bsm[2][B_TILE];
  const int tid  = threadIdx.x;
  const int lane = tid & 31;
  const int wave = tid >> 5;
  const int l15  = lane & 15;
  const int half = lane >> 4;
  const int rowBase = blockIdx.y * 128;
  const int cstart  = blockIdx.x * 1024;

  float rm[2][8], rs[2][8];
#pragma unroll
  for (int r = 0; r < 2; ++r)
#pragma unroll
    for (int v = 0; v < 8; ++v) { rm[r][v] = NEG_BIG; rs[r][v] = 0.0f; }

  for (int ct = 0; ct < 1024; ct += 64) {
    int colBase = cstart + ct;
    if (colBase >= C) break;  // uniform across block

    v8f acc[2][4];
#pragma unroll
    for (int r = 0; r < 2; ++r)
#pragma unroll
      for (int t = 0; t < 4; ++t)
#pragma unroll
        for (int e = 0; e < 8; ++e) acc[r][t][e] = 0.0f;

    if (colBase + 64 <= C)
      kloop<true>(A, B, &Asm[0][0], &Bsm[0][0], rowBase, colBase, K, C, tid, wave, l15, half, acc);
    else
      kloop<false>(A, B, &Asm[0][0], &Bsm[0][0], rowBase, colBase, K, C, tid, wave, l15, half, acc);

    // Fold 64-col tile into running per-row (max, sumexp).
    float bv[4];
    bool  inr[4];
#pragma unroll
    for (int t = 0; t < 4; ++t) {
      int col = colBase + t * 16 + l15;
      inr[t] = (col < C);
      bv[t]  = (bias && inr[t]) ? bias[col] : 0.0f;
    }
#pragma unroll
    for (int r = 0; r < 2; ++r)
#pragma unroll
      for (int v = 0; v < 8; ++v) {
        float val[4];
#pragma unroll
        for (int t = 0; t < 4; ++t)
          val[t] = inr[t] ? (acc[r][t][v] + bv[t]) : NEG_BIG;
        float lm = fmaxf(fmaxf(val[0], val[1]), fmaxf(val[2], val[3]));
        float mNew = fmaxf(rm[r][v], laneMax16(lm));
        float esum = 0.0f;
#pragma unroll
        for (int t = 0; t < 4; ++t) esum += __expf(val[t] - mNew);  // ->0 if masked
        float tsum = laneSum16(esum);
        rs[r][v] = rs[r][v] * __expf(rm[r][v] - mNew) + tsum;
        rm[r][v] = mNew;
      }
  }

  if (l15 == 0) {  // lanes 0 and 16 own 8 rows each (per r)
#pragma unroll
    for (int r = 0; r < 2; ++r)
#pragma unroll
      for (int v = 0; v < 8; ++v) {
        int row = rowBase + wave * 32 + r * 16 + v + 8 * half;
        size_t idx = ((size_t)blockIdx.x * NROWS + row) * 2;
        stats[idx]     = rm[r][v];
        stats[idx + 1] = rs[r][v];
      }
  }
}

// ---------------------------------------------------------------------------
// Per-row target logit: one wave per row.
// mode 0: tail (col = clamp(target-lo, 0, C-1)); mode 1: head (remap).
// ---------------------------------------------------------------------------
__global__ void __launch_bounds__(256)
target_logit(const float* __restrict__ A, const float* __restrict__ B,
             const float* __restrict__ bias, const int* __restrict__ target,
             float* __restrict__ tlog, int K, int C, int lo, int mode) {
  int row  = (int)((blockIdx.x * blockDim.x + threadIdx.x) >> 5);
  int lane = threadIdx.x & 31;
  if (row >= NROWS) return;
  int tgt = target[row];
  int col;
  if (mode == 0) {
    int c = tgt - lo;
    c = c < 0 ? 0 : c;
    c = c > C - 1 ? C - 1 : c;
    col = c;
  } else {
    col = (tgt >= CUT1) ? (CUT0 + 1) : ((tgt >= CUT0) ? CUT0 : tgt);
  }
  float s = 0.0f;
  for (int k = lane; k < K; k += 32)
    s += A[(size_t)row * K + k] * B[(size_t)k * C + col];
#pragma unroll
  for (int m = 16; m >= 1; m >>= 1) s += __shfl_xor(s, m, 32);
  if (lane == 0) tlog[row] = s + (bias ? bias[col] : 0.0f);
}

// ---------------------------------------------------------------------------
// Merge chunk stats per row; nll = -(t - m - log(s)) masked per section.
// ---------------------------------------------------------------------------
__global__ void __launch_bounds__(256)
finalize_section(const float* __restrict__ stats, const float* __restrict__ tlog,
                 const int* __restrict__ target, float* __restrict__ nll,
                 int nchunks, int lo, int hi, int mode) {
  int row = blockIdx.x * blockDim.x + threadIdx.x;
  if (row >= NROWS) return;
  float m = NEG_BIG, s = 0.0f;
  for (int c = 0; c < nchunks; ++c) {
    float cm = stats[((size_t)c * NROWS + row) * 2];
    float cs = stats[((size_t)c * NROWS + row) * 2 + 1];
    float mN = fmaxf(m, cm);
    s = s * __expf(m - mN) + cs * __expf(cm - mN);
    m = mN;
  }
  int tgt = target[row];
  bool mask = (mode == 1) ? true : (tgt >= lo && tgt < hi);
  float v = -(tlog[row] - m - __logf(s));
  nll[row] = mask ? v : 0.0f;
}

// Deterministic final reduction (fixed order, no atomics).
__global__ void __launch_bounds__(256)
final_sum(const float* __restrict__ nll, float* __restrict__ out) {
  __shared__ float red[256];
  float s = 0.0f;
  for (int i = threadIdx.x; i < 3 * NROWS; i += 256) s += nll[i];
  red[threadIdx.x] = s;
  __syncthreads();
  for (int st = 128; st > 0; st >>= 1) {
    if ((int)threadIdx.x < st) red[threadIdx.x] += red[threadIdx.x + st];
    __syncthreads();
  }
  if (threadIdx.x == 0) out[0] = red[0] / (float)NROWS;
}

// ---------------------------------------------------------------------------
extern "C" void kernel_launch(void* const* d_in, const int* in_sizes, int n_in,
                              void* d_out, int out_size, void* d_ws, size_t ws_size,
                              hipStream_t stream) {
  (void)in_sizes; (void)n_in; (void)out_size; (void)ws_size;
  const float* w_in   = (const float*)d_in[0];
  const int*   target = (const int*)d_in[1];
  const float* head_w = (const float*)d_in[2];
  const float* head_b = (const float*)d_in[3];
  const float* t0w1   = (const float*)d_in[4];
  const float* t0w2   = (const float*)d_in[5];
  const float* t1w1   = (const float*)d_in[6];
  const float* t1w2   = (const float*)d_in[7];
  float* out = (float*)d_out;

  // Workspace carve (floats): ~23 MB total.
  float* ws  = (float*)d_ws;
  float* h0  = ws; ws += (size_t)NROWS * 1024;   // w_in @ tail0_w1
  float* h1  = ws; ws += (size_t)NROWS * 256;    // w_in @ tail1_w1
  float* st0 = ws; ws += (size_t)16 * NROWS * 2; // ceil(16000/1024) chunks
  float* st1 = ws; ws += (size_t)30 * NROWS * 2; // ceil(30000/1024)
  float* stH = ws; ws += (size_t)4  * NROWS * 2; // ceil(4002/1024)
  float* tl0 = ws; ws += NROWS;
  float* tl1 = ws; ws += NROWS;
  float* tlH = ws; ws += NROWS;
  float* nll = ws; ws += 3 * NROWS;

  // Bottleneck GEMMs (WG tile 128x64).
  gemm_bf16<<<dim3(1024 / 64, NROWS / 128), 128, 0, stream>>>(w_in, t0w1, h0, NROWS, 1024, 1024);
  gemm_bf16<<<dim3(256 / 64,  NROWS / 128), 128, 0, stream>>>(w_in, t1w1, h1, NROWS, 1024, 256);

  // Streaming logits + softmax stats per section.
  logits_stats<<<dim3(16, NROWS / 128), 128, 0, stream>>>(h0,   t0w2,  nullptr, st0, 1024, 16000);
  logits_stats<<<dim3(30, NROWS / 128), 128, 0, stream>>>(h1,   t1w2,  nullptr, st1, 256,  30000);
  logits_stats<<<dim3(4,  NROWS / 128), 128, 0, stream>>>(w_in, head_w, head_b, stH, 1024, 4002);

  // Gather target logits (one wave / row).
  target_logit<<<(NROWS * 32) / 256, 256, 0, stream>>>(h0,   t0w2,  nullptr, target, tl0, 1024, 16000, CUT0, 0);
  target_logit<<<(NROWS * 32) / 256, 256, 0, stream>>>(h1,   t1w2,  nullptr, target, tl1, 256,  30000, CUT1, 0);
  target_logit<<<(NROWS * 32) / 256, 256, 0, stream>>>(w_in, head_w, head_b, target, tlH, 1024, 4002,  0,    1);

  // Per-row NLL with section masks.
  finalize_section<<<NROWS / 256, 256, 0, stream>>>(st0, tl0, target, nll + 0 * NROWS, 16, CUT0, CUT1, 0);
  finalize_section<<<NROWS / 256, 256, 0, stream>>>(st1, tl1, target, nll + 1 * NROWS, 30, CUT1, CUT2, 0);
  finalize_section<<<NROWS / 256, 256, 0, stream>>>(stH, tlH, target, nll + 2 * NROWS, 4,  0,    0,    1);

  // Deterministic scalar reduce.
  final_sum<<<1, 256, 0, stream>>>(nll, out);
}
